// MinkUNetBase_64785286693655
// MI455X (gfx1250) — matrix operand order repaired and checked by reading, not compiled, not run on hardware
//
#include <hip/hip_runtime.h>
#include <hip/hip_bf16.h>

// CDNA5 (gfx1250) MinkUNet forward.
// Matrix path: v_wmma_f32_16x16x32_bf16 (wave32). Sparse conv = gather rows ->
// WMMA GEMM -> atomic f32 scatter-add. Weights are pre-packed into the exact
// per-lane WMMA B-operand order so the GEMM inner loop does one contiguous
// 32B load per B tile per K-chunk; each wave computes a 16x32 output tile
// (2 accumulators) so every gathered A tile feeds two WMMAs.

typedef __attribute__((ext_vector_type(16))) __bf16 v16bf;
typedef __attribute__((ext_vector_type(8)))  __bf16 v8bf;
typedef __attribute__((ext_vector_type(8)))  float  v8f;

#define BN_EPS 1e-5f

// -------------------------------------------------------------------------
// Elementwise helpers
// -------------------------------------------------------------------------
__global__ __launch_bounds__(256)
void k_cast_bf16(const float* __restrict__ src, __bf16* __restrict__ dst, size_t count) {
  size_t i = (size_t)blockIdx.x * blockDim.x + threadIdx.x;
  if (i < count) dst[i] = (__bf16)src[i];
}

__global__ __launch_bounds__(256)
void k_fill_bias(float* __restrict__ out, const float* __restrict__ b, int C, size_t rows) {
  size_t i = (size_t)blockIdx.x * blockDim.x + threadIdx.x;
  if (i < rows * (size_t)C) out[i] = b[i % C];
}

// -------------------------------------------------------------------------
// Pack W (f32 [K, Cin, Cout]) into per-lane WMMA B-operand order (bf16):
//   out[((k*ntiles + nt)*nch + c)*512 + lane*16 + h]
//     = W[k][c*32 + (lane>>4)*8 + (h&7) + (h>>3)*16][nt*16 + (lane&15)]   (0 outside range)
// so each lane's 16 B halves for one (nt, K-chunk) are contiguous (32B).
// -------------------------------------------------------------------------
__global__ __launch_bounds__(256)
void k_pack_w(const float* __restrict__ W, __bf16* __restrict__ out,
              int K, int Cin, int Cout) {
  const int nch    = (Cin + 31) >> 5;
  const int ntiles = (Cout + 15) >> 4;
  size_t total = (size_t)K * ntiles * nch * 512;
  size_t idx = (size_t)blockIdx.x * blockDim.x + threadIdx.x;
  if (idx >= total) return;
  int    h    = (int)(idx & 15);
  int    lane = (int)((idx >> 4) & 31);
  size_t rest = idx >> 9;                 // ((k*ntiles + nt)*nch + c)
  int    c    = (int)(rest % nch);
  size_t r2   = rest / nch;
  int    nt   = (int)(r2 % ntiles);
  int    k    = (int)(r2 / ntiles);
  int kk  = c * 32 + (lane >> 4) * 8 + (h & 7) + (h >> 3) * 16;
  int col = nt * 16 + (lane & 15);
  __bf16 v = (__bf16)0.0f;
  if (kk < Cin && col < Cout) v = (__bf16)W[((size_t)k * Cin + kk) * Cout + col];
  out[idx] = v;
}

// -------------------------------------------------------------------------
// Gather -> WMMA GEMM -> scatter-add.  One wave per 16x32 output tile
// (2 accumulators sharing one A tile), 4 waves (4 row-tiles) per block.
// gidx/sidx == null => identity rows (dense GEMM: 1x1 downsample, classifier).
// -------------------------------------------------------------------------
__global__ __launch_bounds__(128)
void k_spconv_wmma(const __bf16* __restrict__ xb,   // [*, Cin] bf16 features
                   const __bf16* __restrict__ wp,   // packed B for this offset
                   const int* __restrict__ gidx,    // gather rows (null => identity)
                   const int* __restrict__ sidx,    // scatter rows (null => identity)
                   int mrows,                       // host row bound (exact in gather mode)
                   const int* __restrict__ ndev,    // device row count (identity mode)
                   int Cin, int Cout,
                   float* __restrict__ out)         // [*, Cout] f32 accumulator
{
  const int lane   = threadIdx.x & 31;
  const int mt     = blockIdx.x * 4 + (threadIdx.x >> 5);
  const int nt0    = blockIdx.y * 2;
  const int nt1    = nt0 + 1;
  const int nch    = (Cin + 31) >> 5;
  const int ntiles = (Cout + 15) >> 4;

  int nrows = mrows;
  if (ndev) { int nd = *ndev; nrows = nd < nrows ? nd : nrows; }
  const int row0 = mt * 16;
  if (row0 >= nrows) return;          // wave-uniform: EXEC all-ones at WMMA

  const int  mlane  = lane & 15;
  const int  hiA    = (lane >> 4) * 8;     // A: hi half-wave holds K+8 / K+24
  const int  ia     = row0 + mlane;
  const bool rvalid = ia < nrows;
  const int  grow   = rvalid ? (gidx ? gidx[ia] : ia) : 0;
  const __bf16* __restrict__ arow = xb + (size_t)grow * Cin;

  const bool t1ok = nt1 < ntiles;
  const __bf16* __restrict__ b0p = wp + (size_t)nt0 * nch * 512 + lane * 16;
  const __bf16* __restrict__ b1p = wp + (size_t)(t1ok ? nt1 : nt0) * nch * 512 + lane * 16;

  v8f acc0 = {}, acc1 = {};
  if ((Cin & 31) == 0) {
    for (int c = 0; c < nch; ++c) {
      const int k0 = c * 32;
      v16bf a;
      if (rvalid) {
        v8bf lo = *(const v8bf*)(arow + k0 + hiA);        // 16B contiguous
        v8bf hi = *(const v8bf*)(arow + k0 + hiA + 16);   // 16B contiguous
#pragma unroll
        for (int h = 0; h < 8; ++h) { a[h] = lo[h]; a[h + 8] = hi[h]; }
      } else {
#pragma unroll
        for (int h = 0; h < 16; ++h) a[h] = (__bf16)0.0f;
      }
      v16bf b0 = *(const v16bf*)(b0p + (size_t)c * 512);  // contiguous 32B
      v16bf b1 = *(const v16bf*)(b1p + (size_t)c * 512);
      acc0 = __builtin_amdgcn_wmma_f32_16x16x32_bf16(false, a, false, b0, (short)0, acc0,
                                                     false, false);
      acc1 = __builtin_amdgcn_wmma_f32_16x16x32_bf16(false, a, false, b1, (short)0, acc1,
                                                     false, false);
    }
  } else {
    // ragged Cin (stem Cin==3): guarded scalar A loads, packed B already zero-padded
    for (int c = 0; c < nch; ++c) {
      const int k0 = c * 32;
      v16bf a;
#pragma unroll
      for (int h = 0; h < 8; ++h) {
        int k1 = k0 + hiA + h;
        int k2 = k0 + hiA + 16 + h;
        a[h]     = (rvalid && k1 < Cin) ? arow[k1] : (__bf16)0.0f;
        a[h + 8] = (rvalid && k2 < Cin) ? arow[k2] : (__bf16)0.0f;
      }
      v16bf b0 = *(const v16bf*)(b0p + (size_t)c * 512);
      v16bf b1 = *(const v16bf*)(b1p + (size_t)c * 512);
      acc0 = __builtin_amdgcn_wmma_f32_16x16x32_bf16(false, a, false, b0, (short)0, acc0,
                                                     false, false);
      acc1 = __builtin_amdgcn_wmma_f32_16x16x32_bf16(false, a, false, b1, (short)0, acc1,
                                                     false, false);
    }
  }

  const int col0 = nt0 * 16 + mlane;
  const int col1 = nt1 * 16 + mlane;
#pragma unroll
  for (int v = 0; v < 8; ++v) {
    int io = row0 + (lane >> 4) * 8 + v;   // C/D layout: lane -> N, vgpr -> M
    if (io < nrows) {
      int orow = sidx ? sidx[io] : io;
      float* orp = out + (size_t)orow * Cout;
      if (col0 < Cout)
        __hip_atomic_fetch_add(orp + col0, acc0[v], __ATOMIC_RELAXED,
                               __HIP_MEMORY_SCOPE_AGENT);
      if (t1ok && col1 < Cout)
        __hip_atomic_fetch_add(orp + col1, acc1[v], __ATOMIC_RELAXED,
                               __HIP_MEMORY_SCOPE_AGENT);
    }
  }
}

// -------------------------------------------------------------------------
// BatchNorm: per-channel mean / biased variance over n rows (n read on device)
// -------------------------------------------------------------------------
__global__ __launch_bounds__(256)
void k_bn_stats(const float* __restrict__ y, int C, const int* __restrict__ n_ptr,
                float* __restrict__ mean, float* __restrict__ var) {
  __shared__ float ssum[256];
  __shared__ float ssq[256];
  const int c = blockIdx.x;
  const int n = *n_ptr;
  float sum = 0.f, sq = 0.f;
  for (int i = threadIdx.x; i < n; i += 256) {
    float v = y[(size_t)i * C + c];
    sum += v; sq += v * v;
  }
  ssum[threadIdx.x] = sum; ssq[threadIdx.x] = sq;
  __syncthreads();
  for (int s = 128; s > 0; s >>= 1) {
    if (threadIdx.x < (unsigned)s) {
      ssum[threadIdx.x] += ssum[threadIdx.x + s];
      ssq[threadIdx.x]  += ssq[threadIdx.x + s];
    }
    __syncthreads();
  }
  if (threadIdx.x == 0) {
    float m = ssum[0] / (float)n;
    mean[c] = m;
    var[c]  = ssq[0] / (float)n - m * m;
  }
}

// y -> (y-mean)*rsqrt(var+eps)*g + be  [+ residual] [relu]; writes f32 and/or bf16
__global__ __launch_bounds__(256)
void k_bn_apply(const float* __restrict__ y, const float* __restrict__ mean,
                const float* __restrict__ var, const float* __restrict__ gam,
                const float* __restrict__ bet, const float* __restrict__ resid,
                float* __restrict__ outf, __bf16* __restrict__ outb,
                int C, const int* __restrict__ n_ptr, int relu) {
  size_t idx = (size_t)blockIdx.x * blockDim.x + threadIdx.x;
  size_t tot = (size_t)(*n_ptr) * (size_t)C;
  if (idx >= tot) return;
  int c = (int)(idx % C);
  float v = (y[idx] - mean[c]) * rsqrtf(var[c] + BN_EPS) * gam[c] + bet[c];
  if (resid) v += resid[idx];
  if (relu)  v = fmaxf(v, 0.f);
  if (outf) outf[idx] = v;
  if (outb) outb[idx] = (__bf16)v;
}

__global__ __launch_bounds__(256)
void k_concat(const float* __restrict__ a, int Ca, const float* __restrict__ b, int Cb,
              float* __restrict__ outf, __bf16* __restrict__ outb,
              const int* __restrict__ n_ptr) {
  const int C = Ca + Cb;
  size_t idx = (size_t)blockIdx.x * blockDim.x + threadIdx.x;
  if (idx >= (size_t)(*n_ptr) * (size_t)C) return;
  size_t r = idx / C;
  int    c = (int)(idx % C);
  float v = (c < Ca) ? a[r * (size_t)Ca + c] : b[r * (size_t)Cb + (c - Ca)];
  outf[idx] = v;
  outb[idx] = (__bf16)v;
}

// -------------------------------------------------------------------------
// Host orchestration
// -------------------------------------------------------------------------
extern "C" void kernel_launch(void* const* d_in, const int* in_sizes, int n_in,
                              void* d_out, int out_size, void* d_ws, size_t ws_size,
                              hipStream_t stream) {
  (void)n_in; (void)out_size; (void)ws_size;

  // ---- walk inputs in setup_inputs() insertion order ----
  int cur = 0;
  auto F = [&]() { return (const float*)d_in[cur++]; };
  auto I = [&]() { return (const int*)d_in[cur++]; };

  struct Conv { const float *W, *g, *be, *b; };
  auto rd_conv = [&](bool bias) {
    Conv c; c.W = F(); c.g = F(); c.be = F(); c.b = bias ? F() : nullptr; return c;
  };
  struct Blk { Conv c1, c2; const float *dsW, *dsg, *dsbe; bool has_ds; };
  auto rd_blk = [&](bool ds) {
    Blk b; b.c1 = rd_conv(false); b.c2 = rd_conv(false);
    b.has_ds = ds;
    if (ds) { b.dsW = F(); b.dsg = F(); b.dsbe = F(); }
    else    { b.dsW = b.dsg = b.dsbe = nullptr; }
    return b;
  };

  Conv conv0   = rd_conv(true);
  Conv conv1   = rd_conv(false);
  Blk  block1  = rd_blk(false);       // 32 -> 32, no downsample
  Conv conv2   = rd_conv(false);
  Blk  block2  = rd_blk(true);        // 32 -> 64
  Conv conv3   = rd_conv(false);
  Blk  block3  = rd_blk(true);        // 64 -> 128
  Conv conv4   = rd_conv(false);
  Blk  block4  = rd_blk(true);        // 128 -> 256
  Conv convtr4 = rd_conv(false);
  Blk  block5  = rd_blk(true);        // 384 -> 256
  Conv convtr5 = rd_conv(false);
  Blk  block6  = rd_blk(true);        // 192 -> 128
  Conv convtr6 = rd_conv(false);
  Blk  block7  = rd_blk(true);        // 128 -> 96
  Conv convtr7 = rd_conv(false);
  Blk  block8  = rd_blk(true);        // 128 -> 96
  const float* finW = F();
  const float* finB = F();

  const float* x = F();
  const int n0 = in_sizes[cur - 1] / 3;   // CIN == 3

  // maps["n"]: tuple of 5 ints -> either one 5-elem array or five 1-elem arrays
  const int* d_nlv[5];
  if (in_sizes[cur] == 5) { const int* nn = I(); for (int l = 0; l < 5; ++l) d_nlv[l] = nn + l; }
  else                    { for (int l = 0; l < 5; ++l) d_nlv[l] = I(); }

  struct KM { const int* g[125]; const int* s[125]; int m[125]; int K; };
  auto rd_km = [&](int K) {
    KM km; km.K = K;
    for (int k = 0; k < K; ++k) { km.m[k] = in_sizes[cur]; km.g[k] = I(); km.s[k] = I(); }
    return km;
  };
  KM km0_5 = rd_km(125);
  KM km0_3 = rd_km(27);
  KM km1   = rd_km(27);
  KM km2   = rd_km(27);
  KM km3   = rd_km(27);
  KM km4   = rd_km(27);
  KM d1    = rd_km(8);
  KM d2    = rd_km(8);
  KM d3    = rd_km(8);
  KM d4    = rd_km(8);

  // analytic level-size upper bounds (coords live in (G/2^l)^3 grids)
  const int B0 = n0;
  const int B1 = n0 < 110592 ? n0 : 110592;   // 48^3
  const int B2 = 13824;                        // 24^3
  const int B3 = 1728;                         // 12^3
  const int B4 = 216;                          // 6^3
  const int Bl[5] = {B0, B1, B2, B3, B4};

  // ---- workspace bump allocator ----
  char* wsp = (char*)d_ws;
  size_t off = 0;
  auto alloc = [&](size_t bytes) -> void* {
    off = (off + 255) & ~(size_t)255;
    void* p = wsp + off; off += bytes; return p;
  };
  const size_t GEN = (size_t)B0 * 128;   // covers widest activation at every level

  __bf16* xbf  = (__bf16*)alloc((size_t)B0 * 3   * sizeof(__bf16));
  float*  p1f  = (float*) alloc((size_t)B0 * 32  * sizeof(float));
  __bf16* p1b  = (__bf16*)alloc((size_t)B0 * 32  * sizeof(__bf16));
  float*  b1f  = (float*) alloc((size_t)B1 * 32  * sizeof(float));
  __bf16* b1b  = (__bf16*)alloc((size_t)B1 * 32  * sizeof(__bf16));
  float*  b2f  = (float*) alloc((size_t)B2 * 64  * sizeof(float));
  __bf16* b2b  = (__bf16*)alloc((size_t)B2 * 64  * sizeof(__bf16));
  float*  b3f  = (float*) alloc((size_t)B3 * 128 * sizeof(float));
  __bf16* b3b  = (__bf16*)alloc((size_t)B3 * 128 * sizeof(__bf16));
  float*  curf = (float*) alloc(GEN * sizeof(float));
  __bf16* curb = (__bf16*)alloc(GEN * sizeof(__bf16));
  float*  catf = (float*) alloc(GEN * sizeof(float));
  __bf16* catb = (__bf16*)alloc(GEN * sizeof(__bf16));
  float*  t1f  = (float*) alloc(GEN * sizeof(float));
  __bf16* t1b  = (__bf16*)alloc(GEN * sizeof(__bf16));
  float*  tmpf = (float*) alloc(GEN * sizeof(float));
  float*  idnf = (float*) alloc(GEN * sizeof(float));
  float*  meanb= (float*) alloc(512 * sizeof(float));
  float*  varb = (float*) alloc(512 * sizeof(float));
  // packed weights: max over convs of K * ntiles * nch * 512 (block5.c1)
  __bf16* wbf  = (__bf16*)alloc((size_t)27 * 16 * 12 * 512 * sizeof(__bf16));

  // ---- launch helpers ----
  auto cast_bf = [&](const float* src, __bf16* dst, size_t count) {
    k_cast_bf16<<<dim3((unsigned)((count + 255) / 256)), dim3(256), 0, stream>>>(src, dst, count);
  };
  auto zero = [&](float* p, size_t elems) {
    hipMemsetAsync(p, 0, elems * sizeof(float), stream);
  };
  // pack weights; returns per-offset element stride
  auto packW = [&](const float* W, int K, int Cin, int Cout) -> size_t {
    size_t nch = (size_t)((Cin + 31) / 32), nt = (size_t)((Cout + 15) / 16);
    size_t total = (size_t)K * nt * nch * 512;
    k_pack_w<<<dim3((unsigned)((total + 255) / 256)), dim3(256), 0, stream>>>(
        W, wbf, K, Cin, Cout);
    return nt * nch * 512;
  };
  auto spconv = [&](const __bf16* xb, size_t wstride, int K, int Cin, int Cout,
                    const KM& km, bool transpose, float* outf) {
    for (int k = 0; k < K; ++k) {
      int m = km.m[k];
      if (m <= 0) continue;
      const int* g = transpose ? km.s[k] : km.g[k];
      const int* s = transpose ? km.g[k] : km.s[k];
      dim3 grid((unsigned)((m + 63) / 64), (unsigned)((Cout + 31) / 32));
      k_spconv_wmma<<<grid, dim3(128), 0, stream>>>(
          xb, wbf + (size_t)k * wstride, g, s, m, nullptr, Cin, Cout, outf);
    }
  };
  auto dense = [&](const __bf16* xb, int Cin, int Cout,
                   float* outf, const int* ndev, int nbound) {
    dim3 grid((unsigned)((nbound + 63) / 64), (unsigned)((Cout + 31) / 32));
    k_spconv_wmma<<<grid, dim3(128), 0, stream>>>(
        xb, wbf, nullptr, nullptr, nbound, ndev, Cin, Cout, outf);
  };
  auto bn_stats = [&](const float* y, int C, const int* ndev) {
    k_bn_stats<<<dim3((unsigned)C), dim3(256), 0, stream>>>(y, C, ndev, meanb, varb);
  };
  auto bn_apply = [&](const float* y, const float* g, const float* be, const float* resid,
                      float* outf, __bf16* outb, int C, const int* ndev, int nbound, bool relu) {
    size_t tot = (size_t)nbound * C;
    k_bn_apply<<<dim3((unsigned)((tot + 255) / 256)), dim3(256), 0, stream>>>(
        y, meanb, varb, g, be, resid, outf, outb, C, ndev, relu ? 1 : 0);
  };
  // conv -> BN -> (ReLU).  conv0's bias is a no-op under BN (mean cancels it).
  auto conv_bn = [&](const __bf16* xb, const Conv& c, int K, int Cin, int Cout,
                     const KM& km, bool tr_, const int* ndev, int nbound,
                     float* outf, __bf16* outb, bool relu) {
    size_t ws_ = packW(c.W, K, Cin, Cout);
    zero(tmpf, (size_t)nbound * Cout);
    spconv(xb, ws_, K, Cin, Cout, km, tr_, tmpf);
    bn_stats(tmpf, Cout, ndev);
    bn_apply(tmpf, c.g, c.be, nullptr, outf, outb, Cout, ndev, nbound, relu);
  };
  auto basic_block = [&](const float* xf, const __bf16* xb, const Blk& b, int Ci, int Co,
                         const KM& km, const int* ndev, int nbound,
                         float* outf, __bf16* outb) {
    // t1 = relu(bn(conv3x3(x, c1)))
    conv_bn(xb, b.c1, 27, Ci, Co, km, false, ndev, nbound, t1f, t1b, true);
    // identity path
    const float* resid = xf;
    if (b.has_ds) {
      packW(b.dsW, 1, Ci, Co);
      zero(idnf, (size_t)nbound * Co);
      dense(xb, Ci, Co, idnf, ndev, nbound);
      bn_stats(idnf, Co, ndev);
      bn_apply(idnf, b.dsg, b.dsbe, nullptr, idnf, nullptr, Co, ndev, nbound, false);
      resid = idnf;
    }
    // out = relu(bn(conv3x3(t1, c2)) + resid)
    size_t ws2 = packW(b.c2.W, 27, Co, Co);
    zero(tmpf, (size_t)nbound * Co);
    spconv(t1b, ws2, 27, Co, Co, km, false, tmpf);
    bn_stats(tmpf, Co, ndev);
    bn_apply(tmpf, b.c2.g, b.c2.be, resid, outf, outb, Co, ndev, nbound, true);
  };
  auto concat = [&](const float* a, int Ca, const float* b, int Cb,
                    const int* ndev, int nbound) {
    size_t tot = (size_t)nbound * (Ca + Cb);
    k_concat<<<dim3((unsigned)((tot + 255) / 256)), dim3(256), 0, stream>>>(
        a, Ca, b, Cb, catf, catb, ndev);
  };

  // ---- forward ----
  cast_bf(x, xbf, (size_t)n0 * 3);

  // stem: k=5 conv (125 offsets), Cin=3 -> 32
  conv_bn(xbf, conv0, 125, 3, 32, km0_5, false, d_nlv[0], Bl[0], p1f, p1b, true);

  // encoder
  conv_bn(p1b, conv1, 8, 32, 32, d1, false, d_nlv[1], Bl[1], curf, curb, true);
  basic_block(curf, curb, block1, 32, 32, km1, d_nlv[1], Bl[1], b1f, b1b);
  conv_bn(b1b, conv2, 8, 32, 32, d2, false, d_nlv[2], Bl[2], curf, curb, true);
  basic_block(curf, curb, block2, 32, 64, km2, d_nlv[2], Bl[2], b2f, b2b);
  conv_bn(b2b, conv3, 8, 64, 64, d3, false, d_nlv[3], Bl[3], curf, curb, true);
  basic_block(curf, curb, block3, 64, 128, km3, d_nlv[3], Bl[3], b3f, b3b);
  conv_bn(b3b, conv4, 8, 128, 128, d4, false, d_nlv[4], Bl[4], curf, curb, true);
  basic_block(curf, curb, block4, 128, 256, km4, d_nlv[4], Bl[4], curf, curb);

  // decoder (transposed conv = swapped kernel map)
  conv_bn(curb, convtr4, 8, 256, 256, d4, true, d_nlv[3], Bl[3], curf, curb, true);
  concat(curf, 256, b3f, 128, d_nlv[3], Bl[3]);
  basic_block(catf, catb, block5, 384, 256, km3, d_nlv[3], Bl[3], curf, curb);

  conv_bn(curb, convtr5, 8, 256, 128, d3, true, d_nlv[2], Bl[2], curf, curb, true);
  concat(curf, 128, b2f, 64, d_nlv[2], Bl[2]);
  basic_block(catf, catb, block6, 192, 128, km2, d_nlv[2], Bl[2], curf, curb);

  conv_bn(curb, convtr6, 8, 128, 96, d2, true, d_nlv[1], Bl[1], curf, curb, true);
  concat(curf, 96, b1f, 32, d_nlv[1], Bl[1]);
  basic_block(catf, catb, block7, 128, 96, km1, d_nlv[1], Bl[1], curf, curb);

  conv_bn(curb, convtr7, 8, 96, 96, d1, true, d_nlv[0], Bl[0], curf, curb, true);
  concat(curf, 96, p1f, 32, d_nlv[0], Bl[0]);
  basic_block(catf, catb, block8, 128, 96, km0_3, d_nlv[0], Bl[0], curf, curb);

  // final linear: out = cur @ finW + finB  (96 -> 20, f32 output)
  float* outp = (float*)d_out;
  packW(finW, 1, 96, 20);
  k_fill_bias<<<dim3((unsigned)(((size_t)n0 * 20 + 255) / 256)), dim3(256), 0, stream>>>(
      outp, finB, 20, (size_t)n0);
  dense(curb, 96, 20, outp, d_nlv[0], Bl[0]);
}
